// Discriminator_69063074119966
// MI455X (gfx1250) — compile-verified, matrix operand order checked by reading, not run on hardware
//
#include <hip/hip_runtime.h>
#include <hip/hip_bf16.h>
#include <math.h>

// ---------------- problem constants ----------------
#define GB 64      // batch
#define GT 1024    // seq len
#define GD 128     // input dim
#define GH 512     // hidden

typedef __bf16 bf16_t;
typedef __attribute__((ext_vector_type(16))) __bf16 v16bf;
typedef __attribute__((ext_vector_type(8)))  float  v8f;
typedef __attribute__((ext_vector_type(4)))  unsigned int u32x4;

// ---------------- WMMA helpers ----------------
// 16-bit A-matrix 16x32 (ISA 7.12.2): lanes 0-15 hold K in {0..7,16..23},
// lanes 16-31 hold K in {8..15,24..31}; row M = lane&15. Two b128 loads.
__device__ __forceinline__ v16bf load_frag(const bf16_t* rowk, int half) {
  union { v16bf v; u32x4 q[2]; } f;
  f.q[0] = *(const u32x4*)(rowk + half * 8);
  f.q[1] = *(const u32x4*)(rowk + 16 + half * 8);
  return f.v;
}

__device__ __forceinline__ v8f wmma_bf16(v16bf a, v16bf b, v8f c) {
  return __builtin_amdgcn_wmma_f32_16x16x32_bf16(
      /*neg_a=*/false, a, /*neg_b=*/false, b,
      /*c_mod=*/(short)0, c, /*reuse_a=*/false, /*reuse_b=*/false);
}

__device__ __forceinline__ float sigmoidf_(float x) {
  return 1.0f / (1.0f + __expf(-x));
}

// ---------------- generic WMMA GEMM: C = act(A @ W^T + bias) ----------------
// Register-blocked: each wave owns a 64x16 output slab (4 row tiles) so the
// weight B-fragment is fetched once and reused by 4 WMMAs (load:wmma 5:4).
// A: [M,K] bf16 (optional (s,b)->(b,t) row remap for the [B,T,D] input),
// W: [N,K] bf16 row-major, C: [M,N] bf16. Requires M%64==0, N%16==0, K%32==0.
struct GemmParams {
  const bf16_t* A;
  const bf16_t* W;
  const float*  bias;
  bf16_t*       C;
  int M, N, K, lda, ldw, ldc;
  int act;     // 0 = none, 1 = leaky-relu(0.01)
  int remapT;  // 0 = plain rows; else row m=(s*GB+b) reads A[(b*remapT+t)*lda]
  int rev;     // reverse time (backward direction)
};

__global__ __launch_bounds__(256) void wmma_gemm(GemmParams P) {
  const int lane = threadIdx.x & 31;
  const int wave = blockIdx.x * (blockDim.x >> 5) + (threadIdx.x >> 5);
  const int ntN  = P.N >> 4;
  const int slabs = (P.M >> 6) * ntN;
  if (wave >= slabs) return;
  const int rowG = wave / ntN, colT = wave % ntN;
  const int m0 = rowG << 6, c = colT << 4;
  const int half = lane >> 4, mr = lane & 15;

  const bf16_t* arow[4];
#pragma unroll
  for (int r = 0; r < 4; r++) {
    const int m = m0 + r * 16 + mr;
    if (P.remapT) {
      const int s = m / GB, b = m % GB;
      const int t = P.rev ? (P.remapT - 1 - s) : s;
      arow[r] = P.A + ((size_t)b * P.remapT + t) * P.lda;
    } else {
      arow[r] = P.A + (size_t)m * P.lda;
    }
  }
  const bf16_t* wrow = P.W + (size_t)(c + mr) * P.ldw;

  v8f acc[4] = {};
  for (int k = 0; k < P.K; k += 32) {
    const v16bf bfrag = load_frag(wrow + k, half);
#pragma unroll
    for (int r = 0; r < 4; r++)
      acc[r] = wmma_bf16(load_frag(arow[r] + k, half), bfrag, acc[r]);
  }

  const int j = c + mr;  // C/D layout: N = lane&15, M = 8*half + v
  const float bj = P.bias ? P.bias[j] : 0.0f;
#pragma unroll
  for (int r = 0; r < 4; r++) {
#pragma unroll
    for (int v = 0; v < 8; v++) {
      const int mm = m0 + r * 16 + half * 8 + v;
      float val = acc[r][v] + bj;
      if (P.act == 1) val = (val > 0.0f) ? val : 0.01f * val;
      P.C[(size_t)mm * P.ldc + j] = (bf16_t)val;
    }
  }
}

// ---------------- pipelined GRU step ----------------
// Stage s runs: layer0 (fwd+bwd) at time s, layer1 (fwd+bwd) at time s-1.
// Layer0: one wave owns ALL 4 batch row tiles for its column triple
//   (r,z,n at j, j+H, j+2H): 4 A + 3 B fragments -> 12 WMMAs per k-iter,
//   so Whh is fetched exactly once per step (L2-resident, critical path).
// Layer1: fuses its input projection gi1 = h0_t @ Wih1^T with the recurrence;
//   2-row blocking: 4 A + 6 B fragments -> 12 WMMAs per k-iter.
// Wave map (192 waves): [0,64) layer0 (dir=w>>5, col=w&31),
//                       [64,192) layer1 (dir=(w-64)>>6, rowG, col).
struct StepParams {
  const bf16_t* gi0[2];   // [T*B, 3H] precomputed x@Wih0^T + bih0 (scan order)
  bf16_t*       hseq[2];  // [T*B, H] layer0 hidden sequence
  const bf16_t* whh0[2];  // [3H, H]
  const float*  bhh0[2];
  const bf16_t* wih1[2];  // [3H, H]
  const bf16_t* whh1[2];  // [3H, H]
  const float*  bih1[2];
  const float*  bhh1[2];
  bf16_t*       h1[2];    // ping-pong [2][B*H]
  const bf16_t* zero;     // [B*H] zeros
  int s;
};

__global__ __launch_bounds__(256) void gru_step(StepParams P) {
  const int lane = threadIdx.x & 31;
  const int wave = blockIdx.x * (blockDim.x >> 5) + (threadIdx.x >> 5);
  const int half = lane >> 4, mr = lane & 15;

  if (wave < 64) {
    // ---------------- layer 0 ----------------
    const int dir = wave >> 5;
    const int c = (wave & 31) << 4;
    const int j = c + mr;
    const int t = P.s;
    if (t >= GT) return;
    const bf16_t* hprev = (t == 0) ? P.zero : (P.hseq[dir] + (size_t)(t - 1) * GB * GH);
    const bf16_t* gi    = P.gi0[dir] + (size_t)t * GB * (3 * GH);
    bf16_t*       hout  = P.hseq[dir] + (size_t)t * GB * GH;
    const float*  bhh   = P.bhh0[dir];

    const bf16_t* wR = P.whh0[dir] + (size_t)(c + mr) * GH;
    const bf16_t* wZ = wR + (size_t)GH * GH;
    const bf16_t* wN = wZ + (size_t)GH * GH;

    v8f aR[4] = {}, aZ[4] = {}, aN[4] = {};
    for (int k = 0; k < GH; k += 32) {
      const v16bf bR = load_frag(wR + k, half);
      const v16bf bZ = load_frag(wZ + k, half);
      const v16bf bN = load_frag(wN + k, half);
#pragma unroll
      for (int r = 0; r < 4; r++) {
        const v16bf a = load_frag(hprev + (size_t)(r * 16 + mr) * GH + k, half);
        aR[r] = wmma_bf16(a, bR, aR[r]);
        aZ[r] = wmma_bf16(a, bZ, aZ[r]);
        aN[r] = wmma_bf16(a, bN, aN[r]);
      }
    }
#pragma unroll
    for (int r = 0; r < 4; r++) {
#pragma unroll
      for (int v = 0; v < 8; v++) {
        const int m = r * 16 + half * 8 + v;
        const size_t g = (size_t)m * (3 * GH);
        float rr = sigmoidf_((float)gi[g + j]          + aR[r][v] + bhh[j]);
        float zz = sigmoidf_((float)gi[g + GH + j]     + aZ[r][v] + bhh[GH + j]);
        float nn = tanhf(    (float)gi[g + 2 * GH + j] + rr * (aN[r][v] + bhh[2 * GH + j]));
        float hp = (float)hprev[(size_t)m * GH + j];
        hout[(size_t)m * GH + j] = (bf16_t)((1.0f - zz) * nn + zz * hp);
      }
    }
  } else {
    // ---------------- layer 1 (fused input projection + recurrence) --------
    const int w = wave - 64;
    if (w >= 128) return;
    const int dir = w >> 6;
    const int t6 = w & 63;
    const int m0 = (t6 >> 5) << 5;   // 2 row groups of 32 rows (2 tiles each)
    const int c  = (t6 & 31) << 4;
    const int j = c + mr;
    const int t = P.s - 1;
    if (t < 0) return;
    const bf16_t* xin   = P.hseq[dir] + (size_t)t * GB * GH;
    const bf16_t* hprev = (t == 0) ? P.zero : (P.h1[dir] + (size_t)((t - 1) & 1) * GB * GH);
    bf16_t*       hout  = P.h1[dir] + (size_t)(t & 1) * GB * GH;
    const float*  bi    = P.bih1[dir];
    const float*  bh    = P.bhh1[dir];

    const bf16_t* wiR = P.wih1[dir] + (size_t)(c + mr) * GH;
    const bf16_t* whR = P.whh1[dir] + (size_t)(c + mr) * GH;
    const size_t gstep = (size_t)GH * GH;

    v8f iR[2] = {}, iZ[2] = {}, iN[2] = {}, hR[2] = {}, hZ[2] = {}, hN[2] = {};
    for (int k = 0; k < GH; k += 32) {
      const v16bf biR = load_frag(wiR + k, half);
      const v16bf biZ = load_frag(wiR + gstep + k, half);
      const v16bf biN = load_frag(wiR + 2 * gstep + k, half);
      const v16bf bhR = load_frag(whR + k, half);
      const v16bf bhZ = load_frag(whR + gstep + k, half);
      const v16bf bhN = load_frag(whR + 2 * gstep + k, half);
#pragma unroll
      for (int r = 0; r < 2; r++) {
        const size_t rowoff = (size_t)(m0 + r * 16 + mr) * GH + k;
        const v16bf ax = load_frag(xin + rowoff, half);
        const v16bf ah = load_frag(hprev + rowoff, half);
        iR[r] = wmma_bf16(ax, biR, iR[r]);
        iZ[r] = wmma_bf16(ax, biZ, iZ[r]);
        iN[r] = wmma_bf16(ax, biN, iN[r]);
        hR[r] = wmma_bf16(ah, bhR, hR[r]);
        hZ[r] = wmma_bf16(ah, bhZ, hZ[r]);
        hN[r] = wmma_bf16(ah, bhN, hN[r]);
      }
    }
#pragma unroll
    for (int r = 0; r < 2; r++) {
#pragma unroll
      for (int v = 0; v < 8; v++) {
        const int m = m0 + r * 16 + half * 8 + v;
        float rr = sigmoidf_(iR[r][v] + bi[j]          + hR[r][v] + bh[j]);
        float zz = sigmoidf_(iZ[r][v] + bi[GH + j]     + hZ[r][v] + bh[GH + j]);
        float nn = tanhf(    iN[r][v] + bi[2 * GH + j] + rr * (hN[r][v] + bh[2 * GH + j]));
        float hp = (float)hprev[(size_t)m * GH + j];
        hout[(size_t)m * GH + j] = (bf16_t)((1.0f - zz) * nn + zz * hp);
      }
    }
  }
}

// ---------------- small utility kernels ----------------
__global__ void cvt_f32_bf16(const float* __restrict__ s, bf16_t* __restrict__ d, int n) {
  int i = blockIdx.x * blockDim.x + threadIdx.x;
  if (i < n) d[i] = (bf16_t)s[i];
}

__global__ void zero_bf16(bf16_t* p, int n) {
  int i = blockIdx.x * blockDim.x + threadIdx.x;
  if (i < n) p[i] = (bf16_t)0.0f;
}

// hidden = [h_f_l0 | h_f_l1 | h_b_l0 | h_b_l1] per batch row (bf16)
__global__ void gather_hidden(const bf16_t* hf0, const bf16_t* hf1,
                              const bf16_t* hb0, const bf16_t* hb1,
                              bf16_t* out) {
  int i = blockIdx.x * blockDim.x + threadIdx.x;  // 64*2048
  if (i >= GB * 4 * GH) return;
  int b = i >> 11, j = i & 2047;
  const bf16_t* src = (j < 512)  ? hf0 + b * GH + j
                    : (j < 1024) ? hf1 + b * GH + (j - 512)
                    : (j < 1536) ? hb0 + b * GH + (j - 1024)
                                 : hb1 + b * GH + (j - 1536);
  out[i] = *src;
}

// final critic layer: out[b] = fc1[b,:] . Wfc2 + bfc2   (N=1, plain dot)
__global__ void fc2_dot(const bf16_t* __restrict__ h, const float* __restrict__ w,
                        const float* __restrict__ bias, float* __restrict__ out) {
  int b = threadIdx.x;
  if (b >= GB) return;
  float s = bias[0];
  for (int k = 0; k < GH; k++) s += (float)h[b * GH + k] * w[k];
  out[b] = s;
}

// ---------------- host orchestration ----------------
extern "C" void kernel_launch(void* const* d_in, const int* in_sizes, int n_in,
                              void* d_out, int out_size, void* d_ws, size_t ws_size,
                              hipStream_t stream) {
  (void)in_sizes; (void)n_in; (void)out_size; (void)ws_size;

  const float* x = (const float*)d_in[0];
  // GRU params: f0 @1..4, f1 @5..8, b0 @9..12, b1 @13..16
  const float* Wih[2][2] = {{(const float*)d_in[1],  (const float*)d_in[5]},
                            {(const float*)d_in[9],  (const float*)d_in[13]}};
  const float* Whh[2][2] = {{(const float*)d_in[2],  (const float*)d_in[6]},
                            {(const float*)d_in[10], (const float*)d_in[14]}};
  const float* bih[2][2] = {{(const float*)d_in[3],  (const float*)d_in[7]},
                            {(const float*)d_in[11], (const float*)d_in[15]}};
  const float* bhh[2][2] = {{(const float*)d_in[4],  (const float*)d_in[8]},
                            {(const float*)d_in[12], (const float*)d_in[16]}};
  const float* Wfc0 = (const float*)d_in[17]; const float* bfc0 = (const float*)d_in[18];
  const float* Wfc1 = (const float*)d_in[19]; const float* bfc1 = (const float*)d_in[20];
  const float* Wfc2 = (const float*)d_in[21]; const float* bfc2 = (const float*)d_in[22];
  float* out = (float*)d_out;

  // ---- carve workspace (bf16 elements) ----
  bf16_t* p = (bf16_t*)d_ws;
  auto take = [&](size_t n) { bf16_t* r = p; p += n; return r; };
  bf16_t* xbf   = take((size_t)GB * GT * GD);
  bf16_t* wih0[2], *whh0[2], *wih1[2], *whh1[2];
  for (int d = 0; d < 2; d++) {
    wih0[d] = take((size_t)3 * GH * GD);
    whh0[d] = take((size_t)3 * GH * GH);
    wih1[d] = take((size_t)3 * GH * GH);
    whh1[d] = take((size_t)3 * GH * GH);
  }
  bf16_t* wfc0bf = take((size_t)1024 * 2048);
  bf16_t* wfc1bf = take((size_t)512 * 1024);
  bf16_t* gi0[2]; gi0[0] = take((size_t)GT * GB * 3 * GH); gi0[1] = take((size_t)GT * GB * 3 * GH);
  bf16_t* hseq[2]; hseq[0] = take((size_t)GT * GB * GH); hseq[1] = take((size_t)GT * GB * GH);
  bf16_t* h1[2]; h1[0] = take((size_t)2 * GB * GH); h1[1] = take((size_t)2 * GB * GH);
  bf16_t* zbuf  = take((size_t)GB * GH);
  bf16_t* hid   = take((size_t)GB * 4 * GH);
  bf16_t* fc0o  = take((size_t)GB * 1024);
  bf16_t* fc1o  = take((size_t)GB * 512);

  auto cvt = [&](const float* s, bf16_t* d, int n) {
    cvt_f32_bf16<<<(n + 255) / 256, 256, 0, stream>>>(s, d, n);
  };

  zero_bf16<<<(GB * GH + 255) / 256, 256, 0, stream>>>(zbuf, GB * GH);
  cvt(x, xbf, GB * GT * GD);
  for (int d = 0; d < 2; d++) {
    cvt(Wih[d][0], wih0[d], 3 * GH * GD);
    cvt(Whh[d][0], whh0[d], 3 * GH * GH);
    cvt(Wih[d][1], wih1[d], 3 * GH * GH);
    cvt(Whh[d][1], whh1[d], 3 * GH * GH);
  }
  cvt(Wfc0, wfc0bf, 1024 * 2048);
  cvt(Wfc1, wfc1bf, 512 * 1024);

  // ---- batched input projection: gi0 = x @ Wih0^T + bih0 (both dirs) ----
  auto gemm = [&](GemmParams g) {
    int slabs = (g.M >> 6) * (g.N >> 4);
    wmma_gemm<<<(slabs + 7) / 8, 256, 0, stream>>>(g);
  };
  for (int d = 0; d < 2; d++) {
    GemmParams g{};
    g.A = xbf; g.W = wih0[d]; g.bias = bih[d][0]; g.C = gi0[d];
    g.M = GT * GB; g.N = 3 * GH; g.K = GD;
    g.lda = GD; g.ldw = GD; g.ldc = 3 * GH;
    g.act = 0; g.remapT = GT; g.rev = d;  // backward dir reads time reversed
    gemm(g);
  }

  // ---- pipelined recurrent scan: 1025 stages, 4 GRU cells per stage ----
  StepParams sp{};
  for (int d = 0; d < 2; d++) {
    sp.gi0[d] = gi0[d]; sp.hseq[d] = hseq[d];
    sp.whh0[d] = whh0[d]; sp.bhh0[d] = bhh[d][0];
    sp.wih1[d] = wih1[d]; sp.whh1[d] = whh1[d];
    sp.bih1[d] = bih[d][1]; sp.bhh1[d] = bhh[d][1];
    sp.h1[d] = h1[d];
  }
  sp.zero = zbuf;
  for (int s = 0; s <= GT; s++) {
    sp.s = s;
    gru_step<<<24, 256, 0, stream>>>(sp);  // 192 waves: 64 L0 + 128 L1
  }

  // ---- assemble final hidden state [B, 2*L*H] ----
  gather_hidden<<<(GB * 4 * GH + 255) / 256, 256, 0, stream>>>(
      hseq[0] + (size_t)(GT - 1) * GB * GH, h1[0] + (size_t)((GT - 1) & 1) * GB * GH,
      hseq[1] + (size_t)(GT - 1) * GB * GH, h1[1] + (size_t)((GT - 1) & 1) * GB * GH,
      hid);

  // ---- MLP head ----
  {
    GemmParams g{};
    g.A = hid; g.W = wfc0bf; g.bias = bfc0; g.C = fc0o;
    g.M = GB; g.N = 1024; g.K = 2048; g.lda = 2048; g.ldw = 2048; g.ldc = 1024;
    g.act = 1; g.remapT = 0; g.rev = 0;
    gemm(g);
  }
  {
    GemmParams g{};
    g.A = fc0o; g.W = wfc1bf; g.bias = bfc1; g.C = fc1o;
    g.M = GB; g.N = 512; g.K = 1024; g.lda = 1024; g.ldw = 1024; g.ldc = 512;
    g.act = 1; g.remapT = 0; g.rev = 0;
    gemm(g);
  }
  fc2_dot<<<1, 64, 0, stream>>>(fc1o, Wfc2, bfc2, out);
}
